// MaskContrastModel_35244501631214
// MI455X (gfx1250) — matrix operand order repaired and checked by reading, not compiled, not run on hardware
//
#include <hip/hip_runtime.h>
#include <hip/hip_bf16.h>

// ---------------------------------------------------------------------------
// SupCon loss for MI455X (gfx1250):
//   h = relu(out@W1+b1)@W2+b2 ; f = l2norm(h) ; logits = f f^T / T
//   masked log-softmax -> scalar loss.
// Never materialize the 8192x8192 logits: bf16 features (5 MB, L2-resident) +
// v_wmma_f32_16x16x32_bf16 tiles with fused softmax statistics.
// This revision: dual j-tile per iteration (two independent WMMA accumulator
// chains to hide XDL RAW latency) + one-k-step software pipeline on B loads
// + global_prefetch of the next tile pair.
// ---------------------------------------------------------------------------

#define N_ROWS     8192
#define EMB        300
#define DPAD       320          // padded K (10 wmma k-steps of 32)
#define KSTEPS     10
#define INV_T      10.0f        // 1/TEMP
#define JT_PER_WG  16           // j-tiles handled per wave (8 pairs)
#define NCHUNK     (N_ROWS / 16 / JT_PER_WG)   // 32

typedef __attribute__((ext_vector_type(16))) __bf16          v16bf;
typedef __attribute__((ext_vector_type(8)))  float           v8f;
typedef __attribute__((ext_vector_type(8)))  unsigned short  u16x8;
typedef __attribute__((ext_vector_type(16))) unsigned short  u16x16;

union Frag {
    v16bf  bf;
    u16x16 u;
    struct { u16x8 lo, hi; } h;
};

__device__ __forceinline__ unsigned short f2bf(float f) {
    unsigned u = __float_as_uint(f);
    unsigned r = u + 0x7FFFu + ((u >> 16) & 1u);   // round-to-nearest-even
    return (unsigned short)(r >> 16);
}

// ---------------------------------------------------------------------------
// Kernel 1: projector + L2 normalize -> bf16 features [N_ROWS][DPAD] (zero pad)
// ---------------------------------------------------------------------------
__global__ __launch_bounds__(DPAD)
void proj_kernel(const float* __restrict__ xin,
                 const float* __restrict__ W1, const float* __restrict__ b1,
                 const float* __restrict__ W2, const float* __restrict__ b2,
                 unsigned short* __restrict__ fbf)
{
    __shared__ float xs[EMB];
    __shared__ float t1[EMB];
    __shared__ float ssq;
    const int row = blockIdx.x;
    const int tid = threadIdx.x;

    for (int k = tid; k < EMB; k += DPAD) xs[k] = xin[row * EMB + k];
    if (tid == 0) ssq = 0.0f;
    __syncthreads();

    if (tid < EMB) {
        float acc = b1[tid];
        #pragma unroll 4
        for (int k = 0; k < EMB; ++k) acc = fmaf(xs[k], W1[k * EMB + tid], acc);
        t1[tid] = fmaxf(acc, 0.0f);
    }
    __syncthreads();

    float hval = 0.0f;
    if (tid < EMB) {
        float acc = b2[tid];
        #pragma unroll 4
        for (int k = 0; k < EMB; ++k) acc = fmaf(t1[k], W2[k * EMB + tid], acc);
        hval = acc;
        atomicAdd(&ssq, hval * hval);
    }
    __syncthreads();

    const float scale = 1.0f / fmaxf(sqrtf(ssq), 1e-12f);
    fbf[row * DPAD + tid] = (tid < EMB) ? f2bf(hval * scale) : (unsigned short)0;
}

// ---------------------------------------------------------------------------
// Kernel 2: zero the row statistics (A = sum pos logits, Z = sum exp, Npos)
// ---------------------------------------------------------------------------
__global__ void zero_kernel(float* __restrict__ p, int n)
{
    int i = blockIdx.x * blockDim.x + threadIdx.x;
    if (i < n) p[i] = 0.0f;
}

// ---------------------------------------------------------------------------
// Kernel 3: tiled similarity + fused masked-softmax statistics.
// grid = (512 i-tiles, 32 j-chunks), block = 1 wave (32 lanes).
// A fragments (16x32 bf16, ISA 7.12.2): lane l holds row i_base+(l&15),
//   K = (l>>4)*8 + {0..7} and + {16..23}  -> two 16B loads per k-step.
// B fragments (32x16): lane l holds col j_base+(l&15),
//   K = (l>>4)*8 + {0..7} and + {16..23}  -> two 16B loads per k-step.
// C tile: vgpr r, lane l -> row r + 8*(l>>4), col (l&15).
// Two j-tiles per iteration => two independent accumulator chains.
// ---------------------------------------------------------------------------
__global__ __launch_bounds__(32)
void sim_kernel(const unsigned short* __restrict__ fbf,
                const int* __restrict__ labels,
                float* __restrict__ rowA, float* __restrict__ rowZ,
                float* __restrict__ rowN)
{
    const int lane   = threadIdx.x;
    const int i_base = blockIdx.x * 16;
    const int l15    = lane & 15;
    const int hi     = lane >> 4;          // 0 or 1

    // ---- preload the 10 A fragments for this i-tile (kept in VGPRs) ----
    Frag a[KSTEPS];
    {
        const unsigned short* arow = fbf + (i_base + l15) * DPAD + hi * 8;
        #pragma unroll
        for (int s = 0; s < KSTEPS; ++s) {
            a[s].h.lo = *(const u16x8*)(arow + 32 * s);
            a[s].h.hi = *(const u16x8*)(arow + 32 * s + 16);
        }
    }

    // ---- row labels for the 8 rows this lane owns in the C tile ----
    int rlab[8];
    #pragma unroll
    for (int r = 0; r < 8; ++r) rlab[r] = labels[i_base + r + 8 * hi];

    float sA[8], sZ[8], sN[8];
    #pragma unroll
    for (int r = 0; r < 8; ++r) { sA[r] = 0.0f; sZ[r] = 0.0f; sN[r] = 0.0f; }

    const int jt0 = blockIdx.y * JT_PER_WG;
    for (int jp = 0; jp < JT_PER_WG / 2; ++jp) {
        const int j_base0 = (jt0 + 2 * jp) * 16;
        const int j_base1 = j_base0 + 16;
        const unsigned short* brow0 = fbf + (j_base0 + l15) * DPAD + hi * 8;
        const unsigned short* brow1 = fbf + (j_base1 + l15) * DPAD + hi * 8;

        // speculative warm-up of the next tile pair's rows in WGP$
        __builtin_prefetch(fbf + (j_base1 + 16 + l15) * DPAD, 0, 0);

        // software pipeline: B(step s+1) loads issued before WMMAs of step s
        Frag b0, b1;
        b0.h.lo = *(const u16x8*)(brow0 + 0);
        b0.h.hi = *(const u16x8*)(brow0 + 16);
        b1.h.lo = *(const u16x8*)(brow1 + 0);
        b1.h.hi = *(const u16x8*)(brow1 + 16);

        v8f c0 = {}, c1 = {};
        #pragma unroll
        for (int s = 0; s < KSTEPS; ++s) {
            Frag nb0, nb1;
            if (s + 1 < KSTEPS) {
                nb0.h.lo = *(const u16x8*)(brow0 + 32 * (s + 1));
                nb0.h.hi = *(const u16x8*)(brow0 + 32 * (s + 1) + 16);
                nb1.h.lo = *(const u16x8*)(brow1 + 32 * (s + 1));
                nb1.h.hi = *(const u16x8*)(brow1 + 32 * (s + 1) + 16);
            }
            // two independent accumulator chains interleaved
            c0 = __builtin_amdgcn_wmma_f32_16x16x32_bf16(
                     false, a[s].bf, false, b0.bf, (short)0, c0, false, false);
            c1 = __builtin_amdgcn_wmma_f32_16x16x32_bf16(
                     false, a[s].bf, false, b1.bf, (short)0, c1, false, false);
            if (s + 1 < KSTEPS) { b0 = nb0; b1 = nb1; }
        }

        const int jcol0 = j_base0 + l15;
        const int jcol1 = j_base1 + l15;
        const int clab0 = labels[jcol0];
        const int clab1 = labels[jcol1];

        #pragma unroll
        for (int r = 0; r < 8; ++r) {
            const int irow = i_base + r + 8 * hi;

            const float l0   = c0[r] * INV_T;
            const bool  od0  = (irow != jcol0);
            const bool  p0   = od0 & (rlab[r] == clab0);
            sZ[r] += od0 ? __expf(l0 - INV_T) : 0.0f;   // shift by 1/T (row max)
            sA[r] += p0  ? l0   : 0.0f;
            sN[r] += p0  ? 1.0f : 0.0f;

            const float l1   = c1[r] * INV_T;
            const bool  od1  = (irow != jcol1);
            const bool  p1   = od1 & (rlab[r] == clab1);
            sZ[r] += od1 ? __expf(l1 - INV_T) : 0.0f;
            sA[r] += p1  ? l1   : 0.0f;
            sN[r] += p1  ? 1.0f : 0.0f;
        }
    }

    // ---- reduce across the 16 lanes that share each row, then atomics ----
    #pragma unroll
    for (int r = 0; r < 8; ++r) {
        #pragma unroll
        for (int m = 1; m < 16; m <<= 1) {
            sA[r] += __shfl_xor(sA[r], m, 32);
            sZ[r] += __shfl_xor(sZ[r], m, 32);
            sN[r] += __shfl_xor(sN[r], m, 32);
        }
    }
    if (l15 == 0) {
        #pragma unroll
        for (int r = 0; r < 8; ++r) {
            const int irow = i_base + r + 8 * hi;
            atomicAdd(&rowA[irow], sA[r]);
            atomicAdd(&rowZ[irow], sZ[r]);
            atomicAdd(&rowN[irow], sN[r]);
        }
    }
}

// ---------------------------------------------------------------------------
// Kernel 4: final scalar reduction.
// mean_log_prob_pos_i = (A_i/n_i - 1/T) - log(Z_i)   (shift-invariant softmax)
// loss = -T * sum_valid(...) / max(n_valid, 1)
// ---------------------------------------------------------------------------
__global__ __launch_bounds__(256)
void loss_kernel(const float* __restrict__ rowA, const float* __restrict__ rowZ,
                 const float* __restrict__ rowN, float* __restrict__ outp)
{
    __shared__ float sAcc[256];
    __shared__ float sCnt[256];
    const int t = threadIdx.x;
    float acc = 0.0f, cnt = 0.0f;
    for (int i = t; i < N_ROWS; i += 256) {
        const float n = rowN[i];
        if (n > 0.5f) {
            acc += (rowA[i] / n - INV_T) - logf(rowZ[i]);
            cnt += 1.0f;
        }
    }
    sAcc[t] = acc; sCnt[t] = cnt;
    __syncthreads();
    for (int s = 128; s > 0; s >>= 1) {
        if (t < s) { sAcc[t] += sAcc[t + s]; sCnt[t] += sCnt[t + s]; }
        __syncthreads();
    }
    if (t == 0) outp[0] = -(1.0f / INV_T) * sAcc[0] / fmaxf(sCnt[0], 1.0f);
}

// ---------------------------------------------------------------------------
extern "C" void kernel_launch(void* const* d_in, const int* in_sizes, int n_in,
                              void* d_out, int out_size, void* d_ws, size_t ws_size,
                              hipStream_t stream)
{
    const float* xin    = (const float*)d_in[0];   // [8192,300]
    const int*   labels = (const int*)  d_in[1];   // [8192]
    const float* W1     = (const float*)d_in[2];   // [300,300]
    const float* b1     = (const float*)d_in[3];   // [300]
    const float* W2     = (const float*)d_in[4];   // [300,300]
    const float* b2     = (const float*)d_in[5];   // [300]
    float*       outp   = (float*)d_out;

    char* ws = (char*)d_ws;
    unsigned short* fbf  = (unsigned short*)ws;                       // 8192*320*2 B
    float*          rowA = (float*)(ws + (size_t)N_ROWS * DPAD * 2);  // 3*8192 f32
    float*          rowZ = rowA + N_ROWS;
    float*          rowN = rowZ + N_ROWS;

    // 1) projector + normalize -> bf16 features
    proj_kernel<<<N_ROWS, DPAD, 0, stream>>>(xin, W1, b1, W2, b2, fbf);

    // 2) zero row statistics
    zero_kernel<<<(3 * N_ROWS + 255) / 256, 256, 0, stream>>>(rowA, 3 * N_ROWS);

    // 3) fused similarity WMMA + masked softmax statistics
    dim3 grid(N_ROWS / 16, NCHUNK);
    sim_kernel<<<grid, 32, 0, stream>>>(fbf, labels, rowA, rowZ, rowN);

    // 4) scalar loss
    loss_kernel<<<1, 256, 0, stream>>>(rowA, rowZ, rowN, outp);
}
